// VectorQuantizer_77773267796384
// MI455X (gfx1250) — matrix operand order repaired and checked by reading, not compiled, not run on hardware
//
#include <hip/hip_runtime.h>

typedef __attribute__((ext_vector_type(16))) _Float16 v16h;
typedef __attribute__((ext_vector_type(8)))  float    v8f;

#define NUM_EMB 1024
#define EMB_DIM 64
#define B_SZ    64
#define T_SZ    4096
#define NVEC    (B_SZ * T_SZ)            // 262144 vectors
#define QELEMS  (B_SZ * EMB_DIM * T_SZ)  // 16777216 quantized elems

// workspace layout (bytes)
#define WS_EMBH 0                          // f16 [1024*64] = 131072 B
#define WS_WN   (NUM_EMB * EMB_DIM * 2)    // f32 [1024] half-norms
#define WS_HIST (WS_WN + NUM_EMB * 4)      // f32 [1024] histogram
#define WS_LOSS (WS_HIST + NUM_EMB * 4)    // f32 [1] loss accumulator

// ---------------------------------------------------------------------------
// Prep: f16 codebook, 0.5*||w||^2, zero accumulators (deterministic per call)
// ---------------------------------------------------------------------------
__global__ void __launch_bounds__(256)
vq_prep(const float* __restrict__ emb, _Float16* __restrict__ embh,
        float* __restrict__ wn, float* __restrict__ hist,
        float* __restrict__ lossAcc) {
    int k = blockIdx.x * blockDim.x + threadIdx.x;
    if (k >= NUM_EMB) return;
    const float* w = emb + (size_t)k * EMB_DIM;
    _Float16* oh   = embh + (size_t)k * EMB_DIM;
    float s = 0.f;
#pragma unroll 8
    for (int c = 0; c < EMB_DIM; ++c) {
        float v = w[c];
        s += v * v;
        oh[c] = (_Float16)v;
    }
    wn[k]   = 0.5f * s;
    hist[k] = 0.f;
    if (k == 0) *lossAcc = 0.f;
}

// ---------------------------------------------------------------------------
// Main: one wave per 64-row tile (64 t-values of one batch b).
//   Four 16-row A tiles share every B tile -> 8 WMMAs per B fetch
//   (halves L2 codebook traffic vs M=32). B fragments and the 0.5||w||^2
//   bias for tile n+1 are preloaded before the WMMAs of tile n; the bias is
//   folded into the WMMA C-init (score = x.w - 0.5||w||^2 directly), so no
//   post-WMMA subtract and no load-wait on the critical path.
// ---------------------------------------------------------------------------
__global__ void __launch_bounds__(32)
vq_main(const float* __restrict__ z, const float* __restrict__ emb,
        const _Float16* __restrict__ embh, const float* __restrict__ wn,
        float* __restrict__ hist, float* __restrict__ lossAcc,
        float* __restrict__ out) {
    __shared__ _Float16 tile[64 * 68];   // [t][c] f16, stride 68 (bank-safe)
    __shared__ unsigned idxs[64];

    const int lane = threadIdx.x;
    const int gid  = blockIdx.x;
    const int b    = gid >> 6;           // 64 tiles of 64 rows per batch
    const int t0   = (gid & 63) << 6;
    const float* zb = z + (size_t)b * EMB_DIM * T_SZ + t0;

    // ---- stage z tile (64 t x 64 c) into LDS as f16, transposed to [t][c]
#pragma unroll
    for (int i = 0; i < 2; ++i) {
        int c = lane + 32 * i;
        const float* src = zb + (size_t)c * T_SZ;
#pragma unroll
        for (int j = 0; j < 16; ++j) {
            float4 v = *(const float4*)(src + 4 * j);
            tile[(4 * j + 0) * 68 + c] = (_Float16)v.x;
            tile[(4 * j + 1) * 68 + c] = (_Float16)v.y;
            tile[(4 * j + 2) * 68 + c] = (_Float16)v.z;
            tile[(4 * j + 3) * 68 + c] = (_Float16)v.w;
        }
    }
    __syncthreads();

    // ---- build A fragments for the four 16-row tiles (16-bit A 16x32 layout)
    const int h = lane >> 4;     // half-wave
    const int m = lane & 15;     // A row slot / B column slot
    union F { v16h v; unsigned u[8]; };
    F a0[4], a1[4];
#pragma unroll
    for (int tt = 0; tt < 4; ++tt) {
        const int row = 16 * tt + m;
#pragma unroll
        for (int r = 0; r < 8; ++r) {
            int k = 2 * r + 16 * h;
            a0[tt].u[r] = *(const unsigned*)&tile[row * 68 + k];      // K 0..31
            a1[tt].u[r] = *(const unsigned*)&tile[row * 68 + k + 32]; // K 32..63
        }
    }

    float    bv[4][8];
    unsigned bi[4][8];
#pragma unroll
    for (int tt = 0; tt < 4; ++tt)
#pragma unroll
        for (int r = 0; r < 8; ++r) { bv[tt][r] = -3.4e38f; bi[tt][r] = 0u; }

    // ---- software-pipelined sweep over all 1024 codes, 16 at a time
    const unsigned* eb = (const unsigned*)embh;   // 32 dwords per code row
    F bn0, bn1;
    float wn_next;
    {
        const unsigned* p = eb + (size_t)m * 32;
#pragma unroll
        for (int r = 0; r < 8; ++r) {
            bn0.u[r] = p[r + 8 * h];
            bn1.u[r] = p[r + 8 * h + 16];
        }
        wn_next = wn[m];
    }

#pragma unroll 2
    for (int n0 = 0; n0 < NUM_EMB; n0 += 16) {
        F bc0 = bn0, bc1 = bn1;                 // current B fragments
        const float nwnv = -wn_next;            // current -0.5*||w||^2 bias
        if (n0 + 16 < NUM_EMB) {                // preload next tile's B + bias
            const unsigned* p = eb + (size_t)(n0 + 16 + m) * 32;
#pragma unroll
            for (int r = 0; r < 8; ++r) {
                bn0.u[r] = p[r + 8 * h];
                bn1.u[r] = p[r + 8 * h + 16];
            }
            wn_next = wn[n0 + 16 + m];
        }
        if (n0 + 32 < NUM_EMB)                  // stride-ahead prefetch
            __builtin_prefetch(embh + (size_t)(n0 + 32 + m) * EMB_DIM, 0, 3);

        const unsigned code = (unsigned)(n0 + m);
#pragma unroll
        for (int tt = 0; tt < 4; ++tt) {
            v8f acc;
#pragma unroll
            for (int r = 0; r < 8; ++r) acc[r] = nwnv;  // bias in C-init
            acc = __builtin_amdgcn_wmma_f32_16x16x32_f16(
                      false, a0[tt].v, false, bc0.v, (short)0, acc, false, false);
            acc = __builtin_amdgcn_wmma_f32_16x16x32_f16(
                      false, a1[tt].v, false, bc1.v, (short)0, acc, false, false);
#pragma unroll
            for (int r = 0; r < 8; ++r) {       // D slot: row = r + 8h, col = m
                float s = acc[r];
                if (s > bv[tt][r]) { bv[tt][r] = s; bi[tt][r] = code; }
            }
        }
    }

    // ---- reduce across the 16 column slots (lanes within each half-wave)
#pragma unroll
    for (int off = 1; off < 16; off <<= 1) {
#pragma unroll
        for (int tt = 0; tt < 4; ++tt)
#pragma unroll
            for (int r = 0; r < 8; ++r) {
                float    ov = __shfl_xor(bv[tt][r], off, 32);
                unsigned oi = (unsigned)__shfl_xor((int)bi[tt][r], off, 32);
                if (ov > bv[tt][r] || (ov == bv[tt][r] && oi < bi[tt][r])) {
                    bv[tt][r] = ov; bi[tt][r] = oi;  // first-index tie-break
                }
            }
    }
    if (m == 0) {
#pragma unroll
        for (int tt = 0; tt < 4; ++tt)
#pragma unroll
            for (int r = 0; r < 8; ++r)
                idxs[16 * tt + r + 8 * h] = bi[tt][r];
    }
    __syncthreads();

    // ---- indices output (as float) + histogram: two rows per lane
#pragma unroll
    for (int i = 0; i < 2; ++i) {
        unsigned id = idxs[lane + 32 * i];
        out[(size_t)QELEMS + (size_t)b * T_SZ + t0 + lane + 32 * i] = (float)id;
        atomicAdd(&hist[id], 1.0f);
    }

    // ---- quantized output (STE: q_st == q) + fused MSE accumulation
    float lsum = 0.f;
    float* qb = out + (size_t)b * EMB_DIM * T_SZ + t0;
#pragma unroll 4
    for (int j = 0; j < 128; ++j) {
        int mm = lane + 32 * (j & 1);   // 64 rows x 64 channels
        int c  = j >> 1;
        float q  = emb[(size_t)idxs[mm] * EMB_DIM + c];
        float zv = zb[(size_t)c * T_SZ + mm];
        qb[(size_t)c * T_SZ + mm] = q;
        float d = q - zv;
        lsum += d * d;
    }
#pragma unroll
    for (int off = 16; off >= 1; off >>= 1) lsum += __shfl_xor(lsum, off, 32);
    if (lane == 0) atomicAdd(lossAcc, lsum);
}

// ---------------------------------------------------------------------------
// Finalize: losses + perplexity scalars
// ---------------------------------------------------------------------------
__global__ void __launch_bounds__(256)
vq_final(const float* __restrict__ hist, const float* __restrict__ lossAcc,
         float* __restrict__ out) {
    __shared__ float red[256];
    int t = threadIdx.x;
    float s = 0.f;
    for (int k = t; k < NUM_EMB; k += 256) {
        float p = hist[k] * (1.0f / (float)NVEC);
        s += p * logf(p + 1e-10f);
    }
    red[t] = s;
    __syncthreads();
    for (int off = 128; off >= 1; off >>= 1) {
        if (t < off) red[t] += red[t + off];
        __syncthreads();
    }
    if (t == 0) {
        float mse  = *lossAcc / (float)QELEMS;   // codebook == commitment MSE
        float perp = expf(-red[0]);
        size_t o = (size_t)QELEMS + NVEC;
        out[o + 0] = mse * 1.25f;   // vq_loss = codebook + 0.25*commitment
        out[o + 1] = mse;           // codebook_loss
        out[o + 2] = mse;           // commitment_loss
        out[o + 3] = perp;          // perplexity
    }
}

// ---------------------------------------------------------------------------
extern "C" void kernel_launch(void* const* d_in, const int* in_sizes, int n_in,
                              void* d_out, int out_size, void* d_ws,
                              size_t ws_size, hipStream_t stream) {
    const float* z   = (const float*)d_in[0];   // (64, 64, 4096) fp32
    const float* emb = (const float*)d_in[1];   // (1024, 64) fp32
    float* out = (float*)d_out;

    char* ws = (char*)d_ws;
    _Float16* embh = (_Float16*)(ws + WS_EMBH);
    float*    wn   = (float*)(ws + WS_WN);
    float*    hist = (float*)(ws + WS_HIST);
    float*    loss = (float*)(ws + WS_LOSS);

    vq_prep<<<dim3((NUM_EMB + 255) / 256), dim3(256), 0, stream>>>(
        emb, embh, wn, hist, loss);

    // one wave per 64-row tile: 64 batches * 64 tiles = 4096 blocks of 32
    vq_main<<<dim3(B_SZ * (T_SZ / 64)), dim3(32), 0, stream>>>(
        z, emb, embh, wn, hist, loss, out);

    vq_final<<<dim3(1), dim3(256), 0, stream>>>(hist, loss, out);
}
